// HieraMultiScaleAttention_57939108823143
// MI455X (gfx1250) — matrix operand order repaired and checked by reading, not compiled, not run on hardware
//
#include <hip/hip_runtime.h>
#include <stdint.h>
#include <stddef.h>

typedef __attribute__((ext_vector_type(16))) __bf16 v16bf;
typedef __attribute__((ext_vector_type(8)))  float  v8f;

#define HEADS 12
#define HD    64
#define DIN   384
#define D3    2304
#define DOUT  768
#define BATCH 4
#define S_TOK 4096
#define T_TOK 1024
#define NBH   (BATCH*HEADS)   /* 48 */

// ---------------------------------------------------------------- helpers ----

__device__ __forceinline__ v8f zero8() {
  v8f z;
#pragma unroll
  for (int i = 0; i < 8; ++i) z[i] = 0.0f;
  return z;
}

__device__ __forceinline__ v8f wmma_bf16(v16bf a, v16bf b, v8f c) {
  // D = A(16x32 bf16) * B(32x16 bf16) + C(16x16 f32)
  return __builtin_amdgcn_wmma_f32_16x16x32_bf16(false, a, false, b, (short)0, c,
                                                 false, false);
}

// A fragment (16x32, bf16): lane holds row (lane&15);
// element e -> k = k0 + (e>>3)*16 + (lane>>4)*8 + (e&7)   (per ISA 16-bit A layout)
__device__ __forceinline__ v16bf load_a_bf16(const __bf16* __restrict__ row_ptr,
                                             int lane, int k0) {
  const __bf16* p = row_ptr + k0 + ((lane >> 4) & 1) * 8;
  v16bf a;
#pragma unroll
  for (int blk = 0; blk < 2; ++blk)
#pragma unroll
    for (int i = 0; i < 8; ++i)
      a[blk * 8 + i] = p[blk * 16 + i];
  return a;
}

__device__ __forceinline__ v16bf load_a_f32(const float* __restrict__ row_ptr,
                                            int lane, int k0) {
  const float* p = row_ptr + k0 + ((lane >> 4) & 1) * 8;
  v16bf a;
#pragma unroll
  for (int blk = 0; blk < 2; ++blk)
#pragma unroll
    for (int i = 0; i < 8; ++i)
      a[blk * 8 + i] = (__bf16)p[blk * 16 + i];
  return a;
}

// B fragment (32x16, bf16): lane holds one column of B (= one row of B^T storage);
// element e -> k = k0 + (e>>3)*16 + (lane>>4)*8 + (e&7)
__device__ __forceinline__ v16bf load_b_bf16(const __bf16* __restrict__ col_ptr,
                                             int lane, int k0) {
  const __bf16* p = col_ptr + k0 + ((lane >> 4) & 1) * 8;
  v16bf b;
#pragma unroll
  for (int i = 0; i < 16; ++i) b[i] = p[(i >> 3) * 16 + (i & 7)];
  return b;
}

// ----------------------------------------------------- weight prep kernel ----
// dst[n][k] = (bf16) src[k][n]   (store B^T so B-fragment loads are contiguous)
__global__ void k_transpose_bf16(const float* __restrict__ src,
                                 __bf16* __restrict__ dst, int K, int N) {
  int idx = blockIdx.x * blockDim.x + threadIdx.x;
  if (idx >= K * N) return;
  int n = idx / K;
  int k = idx - n * K;
  dst[idx] = (__bf16)src[(size_t)k * N + n];
}

// ------------------------------------------------------------- QKV GEMM ------
// x (16384 x 384) fp32  @  wqkvT (2304 x 384 bf16, transposed)  + bias
// Scatter results into q[bh][s][64], k[bh][s][64], vT[bh][64][s]  (bf16).
__global__ __launch_bounds__(256) void
k_qkv_gemm(const float* __restrict__ x, const __bf16* __restrict__ wT,
           const float* __restrict__ bias, __bf16* __restrict__ qb,
           __bf16* __restrict__ kb, __bf16* __restrict__ vT) {
  const int lane = threadIdx.x & 31;
  const int wid  = threadIdx.x >> 5;
  const int nl   = lane & 15;
  const int half = (lane >> 4) & 1;
  const int rowM = blockIdx.x * 128 + (wid >> 2) * 64;  // 64 rows per wave
  const int colN = blockIdx.y * 128 + (wid & 3) * 32;   // 32 cols per wave

  v8f acc[4][2];
#pragma unroll
  for (int i = 0; i < 4; ++i)
#pragma unroll
    for (int j = 0; j < 2; ++j) acc[i][j] = zero8();

  for (int k0 = 0; k0 < DIN; k0 += 32) {
    v16bf a[4];
#pragma unroll
    for (int i = 0; i < 4; ++i) {
      const float* rp = x + (size_t)(rowM + i * 16 + nl) * DIN;
      a[i] = load_a_f32(rp, lane, k0);
      __builtin_prefetch(rp + k0 + 32, 0, 1);  // speculative: next K slab
    }
#pragma unroll
    for (int j = 0; j < 2; ++j) {
      v16bf b = load_b_bf16(wT + (size_t)(colN + j * 16 + nl) * DIN, lane, k0);
#pragma unroll
      for (int i = 0; i < 4; ++i) acc[i][j] = wmma_bf16(a[i], b, acc[i][j]);
    }
  }

#pragma unroll
  for (int i = 0; i < 4; ++i)
#pragma unroll
    for (int j = 0; j < 2; ++j) {
      int gn    = colN + j * 16 + nl;
      int which = gn / DOUT;                 // 0=q 1=k 2=v
      int rem   = gn - which * DOUT;
      int head  = rem >> 6;
      int d     = rem & 63;
      float bv  = bias[gn];
#pragma unroll
      for (int r = 0; r < 8; ++r) {
        int gm  = rowM + i * 16 + half * 8 + r;   // global token
        int bi  = gm >> 12;
        int s   = gm & (S_TOK - 1);
        int bh  = bi * HEADS + head;
        __bf16 val = (__bf16)(acc[i][j][r] + bv);
        if (which == 0)
          qb[((size_t)bh * S_TOK + s) * HD + d] = val;
        else if (which == 1)
          kb[((size_t)bh * S_TOK + s) * HD + d] = val;
        else
          vT[((size_t)bh * HD + d) * S_TOK + s] = val;   // pre-transposed V
      }
    }
}

// -------------------------------------------------------- Q pool + scale -----
__global__ void k_pool_q(const __bf16* __restrict__ q, __bf16* __restrict__ qp) {
  int idx = blockIdx.x * blockDim.x + threadIdx.x;  // NBH*T_TOK*HD
  if (idx >= NBH * T_TOK * HD) return;
  int d  = idx & 63;
  int t  = (idx >> 6) & (T_TOK - 1);
  int bh = idx >> 16;
  int ho = t >> 5, wo = t & 31;
  int s00 = (ho * 2) * 64 + (wo * 2);
  const __bf16* base = q + ((size_t)bh * S_TOK) * HD + d;
  float m0 = (float)base[(size_t)s00 * HD];
  float m1 = (float)base[(size_t)(s00 + 1) * HD];
  float m2 = (float)base[(size_t)(s00 + 64) * HD];
  float m3 = (float)base[(size_t)(s00 + 65) * HD];
  float m  = fmaxf(fmaxf(m0, m1), fmaxf(m2, m3));
  qp[idx] = (__bf16)(m * 0.125f);  // fold SCALE = 64^-0.5
}

// --------------------------------------------------- flash attention ---------
// grid (T_TOK/64, NBH), 128 threads (4 waves). Each wave: 16 q-rows, streams all
// 4096 s-tokens in chunks of 64 with online softmax; P routed through LDS to
// convert C-layout -> A-layout for the P@V WMMAs.
__global__ __launch_bounds__(128) void
k_attn(const __bf16* __restrict__ qp, const __bf16* __restrict__ kb,
       const __bf16* __restrict__ vT, __bf16* __restrict__ ao) {
  __shared__ __bf16 lds_p[4][16 * 64];

  const int lane = threadIdx.x & 31;
  const int wid  = threadIdx.x >> 5;
  const int nl   = lane & 15;
  const int half = (lane >> 4) & 1;
  const int bh   = blockIdx.y;
  const int b    = bh / HEADS;
  const int h    = bh - b * HEADS;
  const int t0   = blockIdx.x * 64 + wid * 16;

  const __bf16* qbase = qp + ((size_t)bh * T_TOK) * HD;
  const __bf16* kbase = kb + ((size_t)bh * S_TOK) * HD;
  const __bf16* vbase = vT + ((size_t)bh * HD) * S_TOK;

  v16bf qa0 = load_a_bf16(qbase + (size_t)(t0 + nl) * HD, lane, 0);
  v16bf qa1 = load_a_bf16(qbase + (size_t)(t0 + nl) * HD, lane, 32);

  v8f acc[4];
#pragma unroll
  for (int j = 0; j < 4; ++j) acc[j] = zero8();
  float mrow[8], lrow[8];
#pragma unroll
  for (int r = 0; r < 8; ++r) { mrow[r] = -3.0e38f; lrow[r] = 0.0f; }

  for (int s0 = 0; s0 < S_TOK; s0 += 64) {
    // ---- scores = q @ k^T for a 16x64 tile -------------------------------
    v8f sc[4];
#pragma unroll
    for (int j = 0; j < 4; ++j) {
      const __bf16* kc = kbase + (size_t)(s0 + j * 16 + nl) * HD;
      v16bf b0 = load_b_bf16(kc, lane, 0);
      v16bf b1 = load_b_bf16(kc, lane, 32);
      __builtin_prefetch(kc + 64 * HD, 0, 1);  // speculative: next chunk
      sc[j] = wmma_bf16(qa0, b0, zero8());
      sc[j] = wmma_bf16(qa1, b1, sc[j]);
    }
    // ---- online softmax update ------------------------------------------
    float cmax[8];
#pragma unroll
    for (int r = 0; r < 8; ++r) {
      float m = sc[0][r];
#pragma unroll
      for (int j = 1; j < 4; ++j) m = fmaxf(m, sc[j][r]);
      cmax[r] = m;
    }
#pragma unroll
    for (int off = 8; off >= 1; off >>= 1)
#pragma unroll
      for (int r = 0; r < 8; ++r)
        cmax[r] = fmaxf(cmax[r], __shfl_xor(cmax[r], off, 16));

    float alpha[8], rsum[8];
#pragma unroll
    for (int r = 0; r < 8; ++r) {
      float mnew = fmaxf(mrow[r], cmax[r]);
      alpha[r]   = __expf(mrow[r] - mnew);
      mrow[r]    = mnew;
      rsum[r]    = 0.0f;
    }
#pragma unroll
    for (int j = 0; j < 4; ++j)
#pragma unroll
      for (int r = 0; r < 8; ++r) {
        float p = __expf(sc[j][r] - mrow[r]);
        rsum[r] += p;
        lds_p[wid][(half * 8 + r) * 64 + j * 16 + nl] = (__bf16)p;
      }
#pragma unroll
    for (int off = 8; off >= 1; off >>= 1)
#pragma unroll
      for (int r = 0; r < 8; ++r) rsum[r] += __shfl_xor(rsum[r], off, 16);
#pragma unroll
    for (int r = 0; r < 8; ++r) lrow[r] = lrow[r] * alpha[r] + rsum[r];
#pragma unroll
    for (int j = 0; j < 4; ++j)
#pragma unroll
      for (int r = 0; r < 8; ++r) acc[j][r] *= alpha[r];

    __syncthreads();  // publish P tile (cross-lane) before A-fragment reload
    v16bf pa0 = load_a_bf16(&lds_p[wid][(size_t)nl * 64], lane, 0);
    v16bf pa1 = load_a_bf16(&lds_p[wid][(size_t)nl * 64], lane, 32);

    // ---- acc += P @ V ----------------------------------------------------
#pragma unroll
    for (int j = 0; j < 4; ++j) {
      const __bf16* vc = vbase + (size_t)(j * 16 + nl) * S_TOK;
      v16bf bv0 = load_b_bf16(vc, lane, s0);
      v16bf bv1 = load_b_bf16(vc, lane, s0 + 32);
      __builtin_prefetch(vc + s0 + 64, 0, 1);  // speculative: next chunk
      acc[j] = wmma_bf16(pa0, bv0, acc[j]);
      acc[j] = wmma_bf16(pa1, bv1, acc[j]);
    }
    __syncthreads();  // done reading LDS before next chunk overwrites it
  }

  // ---- normalize + write ao[b][t][h*64+d] (bf16) -------------------------
#pragma unroll
  for (int j = 0; j < 4; ++j)
#pragma unroll
    for (int r = 0; r < 8; ++r) {
      int t   = t0 + half * 8 + r;
      int col = h * HD + j * 16 + nl;
      ao[((size_t)b * T_TOK + t) * DOUT + col] = (__bf16)(acc[j][r] / lrow[r]);
    }
}

// ----------------------------------------------------------- proj GEMM ------
// ao (4096 x 768 bf16) @ wprojT (768 x 768 bf16, transposed) + bias -> fp32 out
__global__ __launch_bounds__(256) void
k_proj_gemm(const __bf16* __restrict__ ao, const __bf16* __restrict__ wT,
            const float* __restrict__ bias, float* __restrict__ out) {
  const int lane = threadIdx.x & 31;
  const int wid  = threadIdx.x >> 5;
  const int nl   = lane & 15;
  const int half = (lane >> 4) & 1;
  const int rowM = blockIdx.x * 128 + (wid >> 2) * 64;
  const int colN = blockIdx.y * 128 + (wid & 3) * 32;

  v8f acc[4][2];
#pragma unroll
  for (int i = 0; i < 4; ++i)
#pragma unroll
    for (int j = 0; j < 2; ++j) acc[i][j] = zero8();

  for (int k0 = 0; k0 < DOUT; k0 += 32) {
    v16bf a[4];
#pragma unroll
    for (int i = 0; i < 4; ++i) {
      const __bf16* rp = ao + (size_t)(rowM + i * 16 + nl) * DOUT;
      a[i] = load_a_bf16(rp, lane, k0);
      __builtin_prefetch(rp + k0 + 32, 0, 1);  // speculative: next K slab
    }
#pragma unroll
    for (int j = 0; j < 2; ++j) {
      v16bf b = load_b_bf16(wT + (size_t)(colN + j * 16 + nl) * DOUT, lane, k0);
#pragma unroll
      for (int i = 0; i < 4; ++i) acc[i][j] = wmma_bf16(a[i], b, acc[i][j]);
    }
  }

#pragma unroll
  for (int i = 0; i < 4; ++i)
#pragma unroll
    for (int j = 0; j < 2; ++j) {
      int gn   = colN + j * 16 + nl;
      float bv = bias[gn];
#pragma unroll
      for (int r = 0; r < 8; ++r) {
        int gm = rowM + i * 16 + half * 8 + r;
        out[(size_t)gm * DOUT + gn] = acc[i][j][r] + bv;
      }
    }
}

// ------------------------------------------------------------- launcher ------
extern "C" void kernel_launch(void* const* d_in, const int* in_sizes, int n_in,
                              void* d_out, int out_size, void* d_ws,
                              size_t ws_size, hipStream_t stream) {
  (void)in_sizes; (void)n_in; (void)out_size; (void)ws_size;
  const float* x      = (const float*)d_in[0];
  const float* w_qkv  = (const float*)d_in[1];
  const float* b_qkv  = (const float*)d_in[2];
  const float* w_proj = (const float*)d_in[3];
  const float* b_proj = (const float*)d_in[4];

  uintptr_t p = (uintptr_t)d_ws;
  auto take = [&](size_t bytes) -> void* {
    p = (p + 255) & ~(uintptr_t)255;
    void* r = (void*)p;
    p += bytes;
    return r;
  };
  __bf16* wqkvT  = (__bf16*)take((size_t)D3 * DIN * 2);
  __bf16* wprojT = (__bf16*)take((size_t)DOUT * DOUT * 2);
  __bf16* qb     = (__bf16*)take((size_t)NBH * S_TOK * HD * 2);
  __bf16* kb     = (__bf16*)take((size_t)NBH * S_TOK * HD * 2);
  __bf16* vT     = (__bf16*)take((size_t)NBH * S_TOK * HD * 2);
  __bf16* qp     = (__bf16*)take((size_t)NBH * T_TOK * HD * 2);
  __bf16* ao     = (__bf16*)take((size_t)BATCH * T_TOK * DOUT * 2);

  k_transpose_bf16<<<(D3 * DIN + 255) / 256, 256, 0, stream>>>(w_qkv, wqkvT,
                                                               DIN, D3);
  k_transpose_bf16<<<(DOUT * DOUT + 255) / 256, 256, 0, stream>>>(w_proj, wprojT,
                                                                  DOUT, DOUT);

  dim3 g1(BATCH * S_TOK / 128, D3 / 128);  // (128, 18)
  k_qkv_gemm<<<g1, 256, 0, stream>>>(x, wqkvT, b_qkv, qb, kb, vT);

  k_pool_q<<<(NBH * T_TOK * HD) / 256, 256, 0, stream>>>(qb, qp);

  dim3 g2(T_TOK / 64, NBH);  // (16, 48)
  k_attn<<<g2, 128, 0, stream>>>(qp, kb, vT, ao);

  dim3 g3(BATCH * T_TOK / 128, DOUT / 128);  // (32, 6)
  k_proj_gemm<<<g3, 256, 0, stream>>>(ao, wprojT, b_proj, (float*)d_out);
}
